// AttentionBlock_89816356094717
// MI455X (gfx1250) — matrix operand order repaired and checked by reading, not compiled, not run on hardware
//
#include <hip/hip_runtime.h>
#include <stdint.h>

typedef __bf16 bf16_t;
typedef __attribute__((ext_vector_type(16))) __bf16 v16bf;
typedef __attribute__((ext_vector_type(8)))  __bf16 v8bf;
typedef __attribute__((ext_vector_type(2)))  __bf16 v2bf;
typedef __attribute__((ext_vector_type(8)))  float  v8f;

__device__ __forceinline__ bf16_t f2bf(float f) { return (bf16_t)f; }

// ---------------------------------------------------------------------------
// Phase 1: 1x1 convs (theta/phi/g) + fused 2x2 maxpool, emit bf16 WMMA operands
//   Qb: [16][4096][8]  (theta, row-major per query)
//   Kb: [16][1024][8]  (pooled phi, row-major per key)
//   Vt: [16][32][1024] (pooled g, TRANSPOSED: channel-major for B-fragments)
// ---------------------------------------------------------------------------
__global__ __launch_bounds__(256) void sagan_proj_pool(
    const float* __restrict__ x,
    const float* __restrict__ w_theta,
    const float* __restrict__ w_phi,
    const float* __restrict__ w_g,
    bf16_t* __restrict__ Qb,
    bf16_t* __restrict__ Kb,
    bf16_t* __restrict__ Vt)
{
  // weights transposed in LDS ([c][o]) so the o-inner loops read contiguous b128s
  __shared__ float sWth[64 * 8];
  __shared__ float sWph[64 * 8];
  __shared__ float sWg [64 * 32];
  const int t = threadIdx.x;
  for (int i = t; i < 64 * 8; i += 256) {
    int o = i & 7, c = i >> 3;
    sWth[i] = w_theta[o * 64 + c];
    sWph[i] = w_phi  [o * 64 + c];
  }
  for (int i = t; i < 64 * 32; i += 256) {
    int o = i & 31, c = i >> 5;
    sWg[i] = w_g[o * 64 + c];
  }
  __syncthreads();

  const int gid = blockIdx.x * 256 + t;          // one thread per pooled position
  const int pw = gid & 31, ph = (gid >> 5) & 31, b = gid >> 10;
  const float* xb = x + (size_t)b * 64 * 4096;

  float phimax[8], gmax[32];
#pragma unroll
  for (int o = 0; o < 8; ++o)  phimax[o] = -3.0e38f;
#pragma unroll
  for (int o = 0; o < 32; ++o) gmax[o] = -3.0e38f;

#pragma unroll
  for (int dy = 0; dy < 2; ++dy)
#pragma unroll
  for (int dx = 0; dx < 2; ++dx) {
    const int hh = 2 * ph + dy, ww = 2 * pw + dx;
    const int pos = hh * 64 + ww;
    float th[8], phv[8], gv[32];
#pragma unroll
    for (int o = 0; o < 8; ++o) { th[o] = 0.f; phv[o] = 0.f; }
#pragma unroll
    for (int o = 0; o < 32; ++o) gv[o] = 0.f;
    for (int c = 0; c < 64; ++c) {
      const float xv = xb[(size_t)c * 4096 + pos];
#pragma unroll
      for (int o = 0; o < 8; ++o) {
        th[o]  += sWth[c * 8 + o] * xv;
        phv[o] += sWph[c * 8 + o] * xv;
      }
#pragma unroll
      for (int o = 0; o < 32; ++o) gv[o] += sWg[c * 32 + o] * xv;
    }
    bf16_t* q = Qb + ((size_t)b * 4096 + pos) * 8;
#pragma unroll
    for (int o = 0; o < 8; ++o) {
      q[o] = f2bf(th[o]);
      phimax[o] = fmaxf(phimax[o], phv[o]);
    }
#pragma unroll
    for (int o = 0; o < 32; ++o) gmax[o] = fmaxf(gmax[o], gv[o]);
  }

  const int pk = ph * 32 + pw;                   // pooled spatial index (s/4)
  bf16_t* kk = Kb + ((size_t)b * 1024 + pk) * 8;
#pragma unroll
  for (int o = 0; o < 8; ++o) kk[o] = f2bf(phimax[o]);
  bf16_t* vt = Vt + (size_t)b * 32 * 1024;
#pragma unroll
  for (int o = 0; o < 32; ++o) vt[(size_t)o * 1024 + pk] = f2bf(gmax[o]);
}

// ---------------------------------------------------------------------------
// Phase 2: fused attention + w_o epilogue + residual.
//  - unnormalized exp softmax (shift-invariant, scores O(16): f32-safe)
//  - denominator via WMMA against all-ones B fragment (no shuffles)
//  - even/odd key split across the two score fragments so each lane's pair of
//    probabilities is key-adjacent: one v_cvt_pk_bf16_f32 + one ds_store_b32
// 512 blocks (16 batches x 32 query-blocks), 8 wave32s/block, 16 queries/wave.
// ---------------------------------------------------------------------------
__global__ __launch_bounds__(256) void sagan_attn(
    const bf16_t* __restrict__ Qb,
    const bf16_t* __restrict__ Kb,
    const bf16_t* __restrict__ Vt,
    const float* __restrict__ w_o,
    const float* __restrict__ gamma_p,
    const float* __restrict__ x,
    float* __restrict__ out)
{
  __shared__ bf16_t sK[1024 * 8];      // whole K for this batch (16 KB)
  __shared__ bf16_t sP[8][16 * 32];    // per-wave P/O staging (8 KB)

  const int t = threadIdx.x;
  const int wave = t >> 5, lane = t & 31;
  const int b = blockIdx.x >> 5;
  const int qbase = (blockIdx.x & 31) * 128 + wave * 16;
  const int lane16 = lane & 15;
  const int laneHi = lane >> 4;

  { // cooperative K stage, 16B granules
    const uint4* src = (const uint4*)(Kb + (size_t)b * 1024 * 8);
    uint4* dst = (uint4*)sK;
    for (int i = t; i < 1024; i += 256) dst[i] = src[i];
  }
  __syncthreads();

  const bf16_t bz = f2bf(0.0f);
  const bf16_t b1 = f2bf(1.0f);

  // Q A-fragment: 16x32, head dim 8 zero-padded to K=32.
  // A layout: lane M=lane16; halfs 0..7 -> K = i + 8*laneHi (only laneHi==0, i<8 real)
  v16bf qa;
#pragma unroll
  for (int i = 0; i < 16; ++i) qa[i] = bz;
  if (laneHi == 0) {
    const v8bf qr = *(const v8bf*)(Qb + ((size_t)b * 4096 + qbase + lane16) * 8);
#pragma unroll
    for (int i = 0; i < 8; ++i) qa[i] = qr[i];
  }

  // all-ones B fragment (row-sum reducer)
  v16bf onesf;
#pragma unroll
  for (int i = 0; i < 16; ++i) onesf[i] = b1;

  // w_o^T B-fragments (4 tiles of 16 output channels, K=32), kept in registers.
  // B layout: lane N=lane16, half i -> K = i + 16*laneHi; B[k][n] = w_o[n][k]
  v16bf wf[4];
#pragma unroll
  for (int tile = 0; tile < 4; ++tile) {
    const float* wr = w_o + (size_t)(tile * 16 + lane16) * 32 + laneHi * 16;
#pragma unroll
    for (int i = 0; i < 16; ++i) wf[tile][i] = f2bf(wr[i]);
  }

  v8f o0, o1, ssum;
#pragma unroll
  for (int r = 0; r < 8; ++r) { o0[r] = 0.f; o1[r] = 0.f; ssum[r] = 0.f; }

  bf16_t* myP = &sP[wave][0];
  const bf16_t* vb = Vt + (size_t)b * 32 * 1024;

  for (int kt = 0; kt < 32; ++kt) {   // 32 keys per step
    const int k32 = kt * 32;

    // Score B-fragments: kb0 <- EVEN keys (k32+2n), kb1 <- ODD keys (k32+2n+1),
    // so column n of (sa,sb) is the key pair (2n, 2n+1). Head dim padded >=8.
    v16bf kb0, kb1;
#pragma unroll
    for (int i = 0; i < 16; ++i) { kb0[i] = bz; kb1[i] = bz; }
    if (laneHi == 0) {
      const v8bf k0 = *(const v8bf*)(sK + (size_t)(k32 + 2 * lane16) * 8);
      const v8bf k1 = *(const v8bf*)(sK + (size_t)(k32 + 2 * lane16 + 1) * 8);
#pragma unroll
      for (int i = 0; i < 8; ++i) { kb0[i] = k0[i]; kb1[i] = k1[i]; }
    }
    v8f zc;
#pragma unroll
    for (int r = 0; r < 8; ++r) zc[r] = 0.f;
    const v8f sa = __builtin_amdgcn_wmma_f32_16x16x32_bf16(false, qa, false, kb0, (short)0, zc, false, false);
    const v8f sb = __builtin_amdgcn_wmma_f32_16x16x32_bf16(false, qa, false, kb1, (short)0, zc, false, false);

    // P = exp(S), packed pairwise (keys 2n,2n+1) -> one b32 store per row
#pragma unroll
    for (int r = 0; r < 8; ++r) {
      const int q = r + 8 * laneHi;
      v2bf pr;
      pr[0] = f2bf(__expf(sa[r]));
      pr[1] = f2bf(__expf(sb[r]));
      *(v2bf*)(myP + q * 32 + 2 * lane16) = pr;   // natural key order in LDS
    }
    // read back as A-fragment (in-order within wave; per-wave LDS region)
    v16bf pf;
    {
      const bf16_t* prow = myP + lane16 * 32;
      const int ko = laneHi * 16;
      const v8bf p0 = *(const v8bf*)(prow + ko);
      const v8bf p1 = *(const v8bf*)(prow + 8 + ko);
#pragma unroll
      for (int i = 0; i < 8; ++i) { pf[i] = p0[i]; pf[8 + i] = p1[i]; }
    }

    // V B-fragments from transposed V: 32B contiguous per lane (L2-resident)
    const v16bf vf0 = *(const v16bf*)(vb + (size_t)lane16 * 1024 + k32 + laneHi * 16);
    const v16bf vf1 = *(const v16bf*)(vb + (size_t)(16 + lane16) * 1024 + k32 + laneHi * 16);

    o0   = __builtin_amdgcn_wmma_f32_16x16x32_bf16(false, pf, false, vf0,   (short)0, o0,   false, false);
    o1   = __builtin_amdgcn_wmma_f32_16x16x32_bf16(false, pf, false, vf1,   (short)0, o1,   false, false);
    ssum = __builtin_amdgcn_wmma_f32_16x16x32_bf16(false, pf, false, onesf, (short)0, ssum, false, false);
  }

  // softmax normalization (denominator replicated across columns by ones-WMMA)
#pragma unroll
  for (int r = 0; r < 8; ++r) {
    const float inv = 1.0f / ssum[r];
    o0[r] *= inv; o1[r] *= inv;
  }

  // O -> LDS -> A-fragment for the w_o epilogue GEMM (once; natural channel order)
#pragma unroll
  for (int r = 0; r < 8; ++r) {
    const int q = r + 8 * laneHi;
    myP[q * 32 + lane16]      = f2bf(o0[r]);
    myP[q * 32 + 16 + lane16] = f2bf(o1[r]);
  }
  v16bf of;
  {
    const bf16_t* prow = myP + lane16 * 32;
    const int ko = laneHi * 16;
    const v8bf p0 = *(const v8bf*)(prow + ko);
    const v8bf p1 = *(const v8bf*)(prow + 8 + ko);
#pragma unroll
    for (int i = 0; i < 8; ++i) { of[i] = p0[i]; of[8 + i] = p1[i]; }
  }

  const float gamma = gamma_p[0];
#pragma unroll
  for (int tile = 0; tile < 4; ++tile) {
    v8f zc;
#pragma unroll
    for (int r = 0; r < 8; ++r) zc[r] = 0.f;
    const v8f d = __builtin_amdgcn_wmma_f32_16x16x32_bf16(false, of, false, wf[tile], (short)0, zc, false, false);
    const int ch = tile * 16 + lane16;
    const int q0 = qbase + laneHi * 8;
    const size_t base = ((size_t)b * 64 + ch) * 4096 + q0;
    const float* xr = x + base;
    float* orow = out + base;
#pragma unroll
    for (int r = 0; r < 8; ++r) orow[r] = gamma * d[r] + xr[r];  // 8 contiguous floats/lane
  }
}

// ---------------------------------------------------------------------------
extern "C" void kernel_launch(void* const* d_in, const int* in_sizes, int n_in,
                              void* d_out, int out_size, void* d_ws, size_t ws_size,
                              hipStream_t stream) {
  const float* x       = (const float*)d_in[0];
  const float* w_theta = (const float*)d_in[1];
  const float* w_phi   = (const float*)d_in[2];
  const float* w_g     = (const float*)d_in[3];
  const float* w_o     = (const float*)d_in[4];
  const float* gamma   = (const float*)d_in[5];
  float* out = (float*)d_out;

  const size_t qBytes = (size_t)16 * 4096 * 8 * sizeof(bf16_t);   // 1 MB
  const size_t kBytes = (size_t)16 * 1024 * 8 * sizeof(bf16_t);   // 256 KB
  bf16_t* Qb = (bf16_t*)d_ws;
  bf16_t* Kb = (bf16_t*)((char*)d_ws + qBytes);
  bf16_t* Vt = (bf16_t*)((char*)d_ws + qBytes + kBytes);

  sagan_proj_pool<<<64, 256, 0, stream>>>(x, w_theta, w_phi, w_g, Qb, Kb, Vt);
  sagan_attn<<<512, 256, 0, stream>>>(Qb, Kb, Vt, w_o, gamma, x, out);
}
